// HybridBranch_29128468201986
// MI455X (gfx1250) — compile-verified, matrix operand order checked by reading, not run on hardware
//
#include <hip/hip_runtime.h>
#include <hip/hip_bf16.h>

#define BB 128
#define TT 32
#define HH 512
#define CC 2400
#define PP 256
#define G4 2048

typedef __attribute__((ext_vector_type(16))) __bf16 v16bf;
typedef __attribute__((ext_vector_type(8)))  __bf16 v8bf;
typedef __attribute__((ext_vector_type(8)))  float  v8f;

// ---------------------------------------------------------------------------
// WMMA bf16 fragment load: operand stored row-major [rows x K], K contiguous.
// ISA 16-bit A/B layout: lane L -> row L%16; elements 0..7 -> K=(L/16)*8..+7,
// elements 8..15 -> K=16+(L/16)*8..+7. Two 16B contiguous loads per fragment
// (compiler merges them into global_load_b128 pairs).
// ---------------------------------------------------------------------------
__device__ __forceinline__ v16bf load_frag_k(const __bf16* __restrict__ base,
                                             int ld, int r, int half, int k0) {
  const __bf16* p = base + (size_t)r * ld + k0 + half * 8;
  v8bf lo = *(const v8bf*)(p);
  v8bf hi = *(const v8bf*)(p + 16);
  v16bf out;
#pragma unroll
  for (int i = 0; i < 8; ++i) { out[i] = lo[i]; out[i + 8] = hi[i]; }
  return out;
}

// One K-sweep of a 16 x (16*NTJ) strip, software-pipelined (double-buffered):
// fragments for k-step i+1 are loaded before the WMMAs of k-step i, so the
// WMMAs only need partial load waits and L2 latency overlaps matrix math.
// KK is compile-time -> fully unrolled, copies renamed away.
template <int NTJ, int KK>
__device__ __forceinline__ void gemm_pass(const __bf16* __restrict__ aBase,
                                          const __bf16* __restrict__ bBase,
                                          int r, int half, v8f (&acc)[NTJ]) {
  v16bf a0 = load_frag_k(aBase, KK, r, half, 0);
  v16bf b0[NTJ];
#pragma unroll
  for (int j = 0; j < NTJ; ++j)
    b0[j] = load_frag_k(bBase + (size_t)j * 16 * KK, KK, r, half, 0);

#pragma unroll
  for (int k0 = 32; k0 < KK; k0 += 32) {
    // prefetch next k-step into fresh registers (live across the WMMAs below)
    v16bf a1 = load_frag_k(aBase, KK, r, half, k0);
    v16bf b1[NTJ];
#pragma unroll
    for (int j = 0; j < NTJ; ++j)
      b1[j] = load_frag_k(bBase + (size_t)j * 16 * KK, KK, r, half, k0);
    // consume current k-step
#pragma unroll
    for (int j = 0; j < NTJ; ++j)
      acc[j] = __builtin_amdgcn_wmma_f32_16x16x32_bf16(false, a0, false, b0[j],
                                                       (short)0, acc[j], false, false);
    a0 = a1;
#pragma unroll
    for (int j = 0; j < NTJ; ++j) b0[j] = b1[j];
  }
#pragma unroll
  for (int j = 0; j < NTJ; ++j)
    acc[j] = __builtin_amdgcn_wmma_f32_16x16x32_bf16(false, a0, false, b0[j],
                                                     (short)0, acc[j], false, false);
}

// C = A[MxK] * Bw[NxK]^T + bias (+ A2 * B2^T if DUAL). One wave per
// 16 x (16*NTJ) strip; A fragment reused across NTJ WMMAs. All shape-control
// is compile-time or wave-uniform -> EXEC stays all-ones around every WMMA.
template <int NTJ, bool DUAL, int KK>
__global__ void k_gemm_bf16(const __bf16* __restrict__ A, const __bf16* __restrict__ Bw,
                            const __bf16* __restrict__ A2, const __bf16* __restrict__ B2,
                            const float* __restrict__ bias, float* __restrict__ Cout,
                            int M, int N, int nBase, int NGroups) {
  const int lane = threadIdx.x & 31;
  const int wave = threadIdx.x >> 5;
  const int MT = M >> 4;
  const int task = blockIdx.x * (blockDim.x >> 5) + wave;
  if (task >= MT * NGroups) return;      // wave-uniform (grids are exact anyway)
  const int m0 = (task / NGroups) << 4;
  const int n0 = nBase + (task % NGroups) * (NTJ * 16);
  const int r = lane & 15, half = lane >> 4;

  v8f acc[NTJ] = {};
  gemm_pass<NTJ, KK>(A + (size_t)m0 * KK, Bw + (size_t)n0 * KK, r, half, acc);
  if (DUAL)
    gemm_pass<NTJ, KK>(A2 + (size_t)m0 * KK, B2 + (size_t)n0 * KK, r, half, acc);

  // C/D layout: lane -> N = lane%16; VGPR rr -> M = (lane/16)*8 + rr
  const int n = lane & 15;
  const int mb = (lane >> 4) * 8;
#pragma unroll
  for (int j = 0; j < NTJ; ++j) {
    const int nc = n0 + 16 * j + n;
    const float bn = bias[nc];
#pragma unroll
    for (int rr = 0; rr < 8; ++rr)
      Cout[(size_t)(m0 + mb + rr) * N + nc] = acc[j][rr] + bn;
  }
}

// ---------------------------------------------------------------------------
// Elementwise / prep kernels
// ---------------------------------------------------------------------------
__global__ void k_f2bf(const float* __restrict__ s, __bf16* __restrict__ d, int n) {
  int i = blockIdx.x * 256 + threadIdx.x;
  if (i < n) d[i] = (__bf16)s[i];
}

__global__ void k_addbias(const float* __restrict__ a, const float* __restrict__ b,
                          float* __restrict__ o, int n) {
  int i = blockIdx.x * 256 + threadIdx.x;
  if (i < n) o[i] = a[i] + b[i];
}

__global__ void k_init(__bf16* __restrict__ hb, float* __restrict__ c) {
  int i = blockIdx.x * 256 + threadIdx.x;   // B*H
  hb[i] = (__bf16)0.0f;
  c[i] = 0.0f;
}

// X[t][b][h] = W_emb[h, text[b,t]] + b_emb[h], stored bf16 (WMMA A operand)
__global__ void k_embed(const float* __restrict__ W_emb, const float* __restrict__ b_emb,
                        const int* __restrict__ text, __bf16* __restrict__ X) {
  size_t idx = (size_t)blockIdx.x * 256 + threadIdx.x;  // T*B*H
  int h = (int)(idx & 511);
  size_t bt = idx >> 9;
  int b = (int)(bt & 127);
  int t = (int)(bt >> 7);
  int cls = text[b * TT + t];
  X[idx] = (__bf16)(W_emb[(size_t)h * CC + cls] + b_emb[h]);
}

__device__ __forceinline__ float sigf(float x) { return 1.0f / (1.0f + __expf(-x)); }

// LSTM1 activation: c1 = sig(f)*c_prev + sig(i)*tanh(g); h1 = sig(o)*tanh(c1)
__global__ void k_lstm1_act(const float* __restrict__ g, const float* __restrict__ c_in,
                            __bf16* __restrict__ h1b) {
  int idx = blockIdx.x * 256 + threadIdx.x;  // B*H
  int b = idx >> 9, j = idx & 511;
  const float* gb = g + (size_t)b * G4;
  float i = sigf(gb[j]);
  float f = sigf(gb[HH + j]);
  float gg = tanhf(gb[2 * HH + j]);
  float o = sigf(gb[3 * HH + j]);
  float c1 = f * c_in[idx] + i * gg;
  h1b[idx] = (__bf16)(o * tanhf(c1));
}

// LSTM2 activation (zero input state): c2 = sig(i)*tanh(g); h2 = sig(o)*tanh(c2)
// (h2, c2) become the carry for the next step.
__global__ void k_lstm2_act(const float* __restrict__ g, float* __restrict__ c_st,
                            __bf16* __restrict__ hb, float* __restrict__ h2f) {
  int idx = blockIdx.x * 256 + threadIdx.x;  // B*H
  int b = idx >> 9, j = idx & 511;
  const float* gb = g + (size_t)b * G4;
  float i = sigf(gb[j]);
  float gg = tanhf(gb[2 * HH + j]);
  float o = sigf(gb[3 * HH + j]);
  float c2 = i * gg;
  c_st[idx] = c2;
  float h2 = o * tanhf(c2);
  hb[idx] = (__bf16)h2;
  h2f[idx] = h2;
}

// ---------------------------------------------------------------------------
// Attention: one block (256 thr) per batch. fmap[b,p,h] == origin[b,h,p].
// scores -> softmax over p -> context[h] = sum_p a[p]*origin[b,h,p]
// ---------------------------------------------------------------------------
__global__ void k_attn(const float* __restrict__ origin, const float* __restrict__ h2,
                       __bf16* __restrict__ ctxb, int t) {
  __shared__ float sh[HH];
  __shared__ float sa[PP];
  __shared__ float red[8];
  const int b = blockIdx.x;
  const int tid = threadIdx.x;
  const int lane = tid & 31, wv = tid >> 5;

  sh[tid] = h2[(size_t)b * HH + tid];
  sh[tid + 256] = h2[(size_t)b * HH + tid + 256];
  __syncthreads();

  const float* fm = origin + (size_t)b * HH * PP;
  float s = 0.0f;
#pragma unroll 4
  for (int h = 0; h < HH; ++h) s = fmaf(fm[(size_t)h * PP + tid], sh[h], s);

  // block max
  float m = s;
#pragma unroll
  for (int o2 = 16; o2; o2 >>= 1) m = fmaxf(m, __shfl_xor(m, o2, 32));
  if (lane == 0) red[wv] = m;
  __syncthreads();
  float bm = red[0];
#pragma unroll
  for (int i = 1; i < 8; ++i) bm = fmaxf(bm, red[i]);
  __syncthreads();

  // block sum of exp
  float e = __expf(s - bm);
  float ssum = e;
#pragma unroll
  for (int o2 = 16; o2; o2 >>= 1) ssum += __shfl_xor(ssum, o2, 32);
  if (lane == 0) red[wv] = ssum;
  __syncthreads();
  float tot = 0.0f;
#pragma unroll
  for (int i = 0; i < 8; ++i) tot += red[i];
  sa[tid] = e / tot;
  __syncthreads();

  // context: wave wv handles h in [wv*64, wv*64+64)
  for (int h = wv * 64; h < wv * 64 + 64; ++h) {
    const float* row = fm + (size_t)h * PP;
    float p = 0.0f;
#pragma unroll
    for (int j = 0; j < 8; ++j) p = fmaf(sa[lane + 32 * j], row[lane + 32 * j], p);
#pragma unroll
    for (int o2 = 16; o2; o2 >>= 1) p += __shfl_xor(p, o2, 32);
    if (lane == 0) ctxb[((size_t)b * TT + t) * HH + h] = (__bf16)p;
  }
}

// ---------------------------------------------------------------------------
extern "C" void kernel_launch(void* const* d_in, const int* in_sizes, int n_in,
                              void* d_out, int out_size, void* d_ws, size_t ws_size,
                              hipStream_t stream) {
  const float* origin = (const float*)d_in[0];
  const int*   text   = (const int*)d_in[1];
  const float* W_emb  = (const float*)d_in[3];
  const float* b_emb  = (const float*)d_in[4];
  const float* W_ih1  = (const float*)d_in[5];
  const float* W_hh1  = (const float*)d_in[6];
  const float* b_ih1  = (const float*)d_in[7];
  const float* b_hh1  = (const float*)d_in[8];
  const float* W_ih2  = (const float*)d_in[9];
  const float* b_ih2  = (const float*)d_in[11];
  const float* b_hh2  = (const float*)d_in[12];
  const float* W_gen  = (const float*)d_in[13];
  const float* b_gen  = (const float*)d_in[14];
  float* out = (float*)d_out;

  char* ws = (char*)d_ws;
  size_t off = 0;
  auto take = [&](size_t bytes) -> char* {
    char* p = ws + off;
    off += bytes;
    off = (off + 255) & ~(size_t)255;
    return p;
  };
  __bf16* Xemb  = (__bf16*)take((size_t)TT * BB * HH * 2);
  __bf16* Wih1b = (__bf16*)take((size_t)G4 * HH * 2);
  __bf16* Whh1b = (__bf16*)take((size_t)G4 * HH * 2);
  __bf16* Wih2b = (__bf16*)take((size_t)G4 * HH * 2);
  __bf16* Wgenb = (__bf16*)take((size_t)CC * HH * 2);
  float*  bias1 = (float*)take((size_t)G4 * 4);
  float*  bias2 = (float*)take((size_t)G4 * 4);
  __bf16* h_bf  = (__bf16*)take((size_t)BB * HH * 2);
  __bf16* h1_bf = (__bf16*)take((size_t)BB * HH * 2);
  float*  c_st  = (float*)take((size_t)BB * HH * 4);
  float*  h2f   = (float*)take((size_t)BB * HH * 4);
  float*  gates = (float*)take((size_t)BB * G4 * 4);
  __bf16* ctxb  = (__bf16*)take((size_t)BB * TT * HH * 2);

  // --- prologue: bf16 weight conversion (L2-resident thereafter), biases, embed
  k_f2bf<<<4096, 256, 0, stream>>>(W_ih1, Wih1b, G4 * HH);
  k_f2bf<<<4096, 256, 0, stream>>>(W_hh1, Whh1b, G4 * HH);
  k_f2bf<<<4096, 256, 0, stream>>>(W_ih2, Wih2b, G4 * HH);
  k_f2bf<<<4800, 256, 0, stream>>>(W_gen, Wgenb, CC * HH);
  k_addbias<<<8, 256, 0, stream>>>(b_ih1, b_hh1, bias1, G4);
  k_addbias<<<8, 256, 0, stream>>>(b_ih2, b_hh2, bias2, G4);
  k_embed<<<(TT * BB * HH) / 256, 256, 0, stream>>>(W_emb, b_emb, text, Xemb);
  k_init<<<(BB * HH) / 256, 256, 0, stream>>>(h_bf, c_st);

  // gate GEMMs: M=128 (8 M-tiles), N=2048 -> 32 groups of 4 tiles -> 256 tasks
  const int gate_blocks = (8 * 32) / 8;            // 32 blocks * 8 waves
  // epilogue: M=4096 (256 M-tiles); N=2400 -> 37 groups of 4 (n<2368) + 1 of 2
  const int out_blocks4 = (256 * 37) / 8;          // 1184 blocks
  const int out_blocks2 = (256 * 1) / 8;           // 32 blocks

  // --- recurrent chain: 32 sequential steps
  for (int t = 0; t < TT; ++t) {
    // gates1 = x_emb_t @ Wih1^T + h @ Whh1^T + (b_ih1+b_hh1)
    k_gemm_bf16<4, true, HH><<<gate_blocks, 256, 0, stream>>>(
        Xemb + (size_t)t * BB * HH, Wih1b, h_bf, Whh1b, bias1, gates,
        BB, G4, 0, 32);
    k_lstm1_act<<<(BB * HH) / 256, 256, 0, stream>>>(gates, c_st, h1_bf);
    // gates2 = h1 @ Wih2^T + (b_ih2+b_hh2)   (zero hidden state -> no W_hh2 term)
    k_gemm_bf16<4, false, HH><<<gate_blocks, 256, 0, stream>>>(
        h1_bf, Wih2b, nullptr, nullptr, bias2, gates, BB, G4, 0, 32);
    k_lstm2_act<<<(BB * HH) / 256, 256, 0, stream>>>(gates, c_st, h_bf, h2f);
    k_attn<<<BB, 256, 0, stream>>>(origin, h2f, ctxb, t);
  }

  // --- epilogue: g = ctx @ W_gen^T + b_gen   [4096 x 2400]
  k_gemm_bf16<4, false, HH><<<out_blocks4, 256, 0, stream>>>(
      ctxb, Wgenb, nullptr, nullptr, b_gen, out, BB * TT, CC, 0, 37);
  k_gemm_bf16<2, false, HH><<<out_blocks2, 256, 0, stream>>>(
      ctxb, Wgenb, nullptr, nullptr, b_gen, out, BB * TT, CC, 2368, 1);
}
